// DeepRAPM_14688788152509
// MI455X (gfx1250) — matrix operand order repaired and checked by reading, not compiled, not run on hardware
//
#include <hip/hip_runtime.h>
#include <hip/hip_bf16.h>
#include <math.h>

typedef __attribute__((ext_vector_type(16))) _Float16 v16h;
typedef __attribute__((ext_vector_type(8)))  float    v8f;

#define BB    8192
#define NPL   5000
#define DIM   512
#define NHD   8
#define NLAY  4
#define NCL   5
#define DHD   64
#define FFD   2048
#define NTOK  (BB * 10)   // 81920

// Flip to 0 if the assembler rejects the async-to-LDS mnemonic.
#define USE_ASYNC_LDS 1

// ---- GEMM tiling ----
#define BM 128
#define BN 128
#define BK 32
#define LDA_S 48   // LDS row stride (halves) for A tile  (16B-aligned rows, depad banks)
#define LDB_S 48   // LDS row stride (halves) for B^T tile

// ---- workspace layout (offsets in halves) ----
static constexpr size_t O_PLAYER = 0;
static constexpr size_t O_ROLE   = O_PLAYER + (size_t)NPL * DIM;
static constexpr size_t O_PW1    = O_ROLE   + (size_t)2 * DIM;
static constexpr size_t O_PW2    = O_PW1    + (size_t)1024 * 1024;
static constexpr size_t O_WQ     = O_PW2    + (size_t)1024 * 512;
static constexpr size_t O_WK     = O_WQ     + (size_t)NLAY * DIM * DIM;
static constexpr size_t O_WV     = O_WK     + (size_t)NLAY * DIM * DIM;
static constexpr size_t O_WO     = O_WV     + (size_t)NLAY * DIM * DIM;
static constexpr size_t O_F1     = O_WO     + (size_t)NLAY * DIM * DIM;
static constexpr size_t O_F2     = O_F1     + (size_t)NLAY * DIM * FFD;
static constexpr size_t O_HW1    = O_F2     + (size_t)NLAY * FFD * DIM;
static constexpr size_t O_Z      = O_HW1    + (size_t)DIM * DIM;
static constexpr size_t O_ZN     = O_Z      + (size_t)NTOK * DIM;   // also reused as attention O
static constexpr size_t O_Q      = O_ZN     + (size_t)NTOK * DIM;   // also proj-hidden (with O_K) and FFN hidden
static constexpr size_t O_KB     = O_Q      + (size_t)NTOK * DIM;
static constexpr size_t O_VB     = O_KB     + (size_t)NTOK * DIM;
static constexpr size_t O_ZP     = O_VB     + (size_t)NTOK * DIM;
static constexpr size_t O_H1     = O_ZP     + (size_t)BB * DIM;
static constexpr size_t HALF_TOTAL = O_H1   + (size_t)BB * DIM;
static constexpr size_t WS_NEED  = HALF_TOTAL * 2 + (size_t)2 * NTOK * 4;

// ================= elementwise f32 -> f16 =================
__global__ __launch_bounds__(256) void cvt_k(const float* __restrict__ in,
                                             _Float16* __restrict__ out, int n) {
    int i = blockIdx.x * 256 + threadIdx.x;
    if (i < n) out[i] = (_Float16)in[i];
}

// ================= per-token id prep =================
__global__ __launch_bounds__(256) void prep_k(const int* __restrict__ off_ids,
                                              const int* __restrict__ def_ids,
                                              const int* __restrict__ off_pos,
                                              const int* __restrict__ def_pos,
                                              int* __restrict__ pids,
                                              int* __restrict__ cids) {
    int row = blockIdx.x * 256 + threadIdx.x;
    if (row >= NTOK) return;
    int b = row / 10, t = row % 10;
    int pid, pos, role;
    if (t < 5) { pid = off_ids[b * 5 + t];     pos = off_pos[b * 5 + t];     role = 0; }
    else       { pid = def_ids[b * 5 + t - 5]; pos = def_pos[b * 5 + t - 5]; role = 1; }
    pids[row] = pid;
    cids[row] = role * 5 + pos;
}

// ================= WMMA helpers =================
__device__ __forceinline__ v8f wmma_f16(v16h a, v16h b, v8f c) {
    return __builtin_amdgcn_wmma_f32_16x16x32_f16(false, a, false, b, (short)0, c,
                                                  false, false);
}

// A fragment (16x32 f16): lane m=lane&15, g=lane>>4; k runs {8g..8g+7} and {8g+16..8g+23}
__device__ __forceinline__ v16h load_a_frag(const _Float16* base, int lane) {
    int m = lane & 15, g = lane >> 4;
    const _Float16* p = base + (size_t)m * LDA_S + g * 8;
    union { v16h h; uint4 q[2]; } u;
    u.q[0] = *(const uint4*)(p);
    u.q[1] = *(const uint4*)(p + 16);
    return u.h;
}

// B fragment (32x16 f16), LDS tile stored transposed [n][k]: lane n=lane&15, k runs {8g..8g+7} and {8g+16..8g+23}
__device__ __forceinline__ v16h load_b_frag(const _Float16* base, int lane) {
    int n = lane & 15, g = lane >> 4;
    const _Float16* p = base + (size_t)n * LDB_S + g * 8;
    union { v16h h; uint4 q[2]; } u;
    u.q[0] = *(const uint4*)(p);
    u.q[1] = *(const uint4*)(p + 16);
    return u.h;
}

// Async global->LDS b128 copy (CDNA5 GLOBAL_LOAD_ASYNC_TO_LDS_B128, ASYNCcnt-tracked).
// Low 32 bits of a generic LDS pointer are the LDS byte offset (aperture mapping).
__device__ __forceinline__ void async_copy_b128(void* lds_dst, const void* gsrc) {
#if USE_ASYNC_LDS && defined(__HIP_DEVICE_COMPILE__)
    unsigned int       l = (unsigned int)(unsigned long long)lds_dst;
    unsigned long long g = (unsigned long long)gsrc;
    asm volatile("global_load_async_to_lds_b128 %0, %1, off"
                 :: "v"(l), "v"(g) : "memory");
#else
    *(uint4*)lds_dst = *(const uint4*)gsrc;
#endif
}

__device__ __forceinline__ void async_wait_all() {
#if USE_ASYNC_LDS && defined(__HIP_DEVICE_COMPILE__)
    asm volatile("s_wait_asynccnt 0x0" ::: "memory");
#endif
}

// ================= tiled WMMA GEMM  C = [resid +] act(A @ W + bias) =================
// A: MxK f16 (lda), W: KxN f16 row-major (ldw), C: MxN f16 (ldc). All dims divide tiles.
// Block 256 threads = 8 waves; block tile 128x128; wave tile 32x64 (2x4 WMMA subtiles).
template <bool GATHER>
__global__ __launch_bounds__(256) void gemm_f16_k(
    const _Float16* __restrict__ A, int lda,
    const _Float16* __restrict__ W, int ldw,
    const float*    __restrict__ bias,
    const _Float16* __restrict__ resid,
    _Float16*       __restrict__ C, int ldc,
    int K, int relu,
    const int*      __restrict__ pids,
    const _Float16* __restrict__ playerTab,
    const _Float16* __restrict__ roleTab)
{
    __shared__ __align__(16) _Float16 As[BM * LDA_S];
    __shared__ __align__(16) _Float16 Bs[BN * LDB_S];

    const int tid  = threadIdx.x;
    const int lane = tid & 31;
    const int wave = tid >> 5;
    const int bm0  = blockIdx.y * BM;
    const int bn0  = blockIdx.x * BN;
    const int wm   = (wave >> 1) * 32;  // wave row origin in tile (0,32,64,96)
    const int wn   = (wave & 1) * 64;   // wave col origin in tile (0,64)

    v8f acc[2][4] = {};

    const int ar = tid >> 2;            // 0..63
    const int ac = (tid & 3) * 8;       // 0,8,16,24
    const int bk = tid >> 3;            // 0..31
    const int bn = (tid & 7) * 8;       // 0..56

    for (int k0 = 0; k0 < K; k0 += BK) {
        // ---- stage A tile (128x32) ----
        if constexpr (!GATHER) {
            async_copy_b128(&As[(size_t)ar * LDA_S + ac],
                            A + (size_t)(bm0 + ar) * lda + k0 + ac);
            async_copy_b128(&As[(size_t)(ar + 64) * LDA_S + ac],
                            A + (size_t)(bm0 + ar + 64) * lda + k0 + ac);
            if (k0 + BK < K)
                __builtin_prefetch(A + (size_t)(bm0 + ar) * lda + k0 + BK + ac, 0, 0);
        } else {
            int kk = k0 + ac;
            #pragma unroll
            for (int half = 0; half < 2; ++half) {
                int rr  = ar + half * 64;
                int row = bm0 + rr;
                const _Float16* src;
                if (kk < 512) src = playerTab + (size_t)pids[row] * 512 + kk;
                else {
                    int role = ((row % 10) >= 5) ? 1 : 0;
                    src = roleTab + (size_t)role * 512 + (kk - 512);
                }
                *(uint4*)&As[(size_t)rr * LDA_S + ac] = *(const uint4*)src;
            }
        }
        // ---- stage B tile (32x128), transposed into [n][k] ----
        #pragma unroll
        for (int rep = 0; rep < 2; ++rep) {
            int nn2 = bn + rep * 64;
            union { uint4 q; _Float16 h[8]; } u;
            u.q = *(const uint4*)(W + (size_t)(k0 + bk) * ldw + bn0 + nn2);
            #pragma unroll
            for (int i = 0; i < 8; ++i)
                Bs[(size_t)(nn2 + i) * LDB_S + bk] = u.h[i];
        }
        if (k0 + BK < K)
            __builtin_prefetch(W + (size_t)(k0 + BK + bk) * ldw + bn0 + bn, 0, 0);

        async_wait_all();
        __syncthreads();

        v16h af[2], bf[4];
        #pragma unroll
        for (int i = 0; i < 2; ++i)
            af[i] = load_a_frag(&As[(size_t)(wm + i * 16) * LDA_S], lane);
        #pragma unroll
        for (int j = 0; j < 4; ++j)
            bf[j] = load_b_frag(&Bs[(size_t)(wn + j * 16) * LDB_S], lane);
        #pragma unroll
        for (int i = 0; i < 2; ++i)
            #pragma unroll
            for (int j = 0; j < 4; ++j)
                acc[i][j] = wmma_f16(af[i], bf[j], acc[i][j]);
        __syncthreads();
    }

    // ---- epilogue: bias + relu + residual, f16 store ----
    const int nn = lane & 15, g = lane >> 4;
    #pragma unroll
    for (int i = 0; i < 2; ++i) {
        #pragma unroll
        for (int j = 0; j < 4; ++j) {
            #pragma unroll
            for (int e = 0; e < 8; ++e) {
                int m   = bm0 + wm + i * 16 + g * 8 + e;
                int col = bn0 + wn + j * 16 + nn;
                float v = acc[i][j][e];
                if (bias)  v += bias[col];
                if (relu)  v  = fmaxf(v, 0.0f);
                if (resid) v += (float)resid[(size_t)m * ldc + col];
                C[(size_t)m * ldc + col] = (_Float16)v;
            }
        }
    }
}

// ================= LayerNorm (wave per row, D=512) =================
__global__ __launch_bounds__(256) void ln_k(const _Float16* __restrict__ X,
                                            const float* __restrict__ gam,
                                            const float* __restrict__ bet,
                                            _Float16* __restrict__ Y) {
    int lane = threadIdx.x & 31, wave = threadIdx.x >> 5;
    int row  = blockIdx.x * 8 + wave;
    if (row >= NTOK) return;
    const _Float16* x = X + (size_t)row * DIM;
    float v[16], s = 0.f;
    #pragma unroll
    for (int t = 0; t < 16; ++t) { v[t] = (float)x[lane + t * 32]; s += v[t]; }
    #pragma unroll
    for (int o = 16; o > 0; o >>= 1) s += __shfl_xor(s, o, 32);
    float mu = s * (1.0f / 512.0f);
    float var = 0.f;
    #pragma unroll
    for (int t = 0; t < 16; ++t) { float d = v[t] - mu; var += d * d; }
    #pragma unroll
    for (int o = 16; o > 0; o >>= 1) var += __shfl_xor(var, o, 32);
    float rs = rsqrtf(var * (1.0f / 512.0f) + 1e-5f);
    _Float16* y = Y + (size_t)row * DIM;
    #pragma unroll
    for (int t = 0; t < 16; ++t) {
        int c = lane + t * 32;
        y[c] = (_Float16)((v[t] - mu) * rs * gam[c] + bet[c]);
    }
}

// ================= attention core (thread per (b,h,i)) =================
__device__ __forceinline__ void load64f(const _Float16* p, float* dst) {
    const uint4* q = (const uint4*)p;
    #pragma unroll
    for (int i = 0; i < 8; ++i) {
        union { uint4 u; _Float16 h[8]; } t; t.u = q[i];
        #pragma unroll
        for (int j = 0; j < 8; ++j) dst[i * 8 + j] = (float)t.h[j];
    }
}

__global__ __launch_bounds__(256) void attn_k(const _Float16* __restrict__ Q,
                                              const _Float16* __restrict__ Kt,
                                              const _Float16* __restrict__ V,
                                              const float* __restrict__ biasB, // [H][10][10]
                                              const int* __restrict__ cids,
                                              _Float16* __restrict__ O) {
    int idx = blockIdx.x * 256 + threadIdx.x;       // (b*8 + h)*10 + i
    if (idx >= BB * NHD * 10) return;
    int i = idx % 10, bh = idx / 10;
    int h = bh % NHD, b = bh / NHD;
    int row0 = b * 10;

    float q[DHD];
    load64f(Q + ((size_t)(row0 + i) * DIM + h * DHD), q);
    int ci = cids[row0 + i];

    float s[10], mx = -1e30f;
    for (int j = 0; j < 10; ++j) {
        float kv[DHD];
        load64f(Kt + ((size_t)(row0 + j) * DIM + h * DHD), kv);
        float d = 0.f;
        #pragma unroll
        for (int t = 0; t < DHD; ++t) d += q[t] * kv[t];
        s[j] = d * 0.125f + biasB[(h * 10 + ci) * 10 + cids[row0 + j]];
        mx = fmaxf(mx, s[j]);
    }
    float denom = 0.f;
    for (int j = 0; j < 10; ++j) { s[j] = __expf(s[j] - mx); denom += s[j]; }
    float inv = 1.0f / denom;

    float o[DHD];
    #pragma unroll
    for (int t = 0; t < DHD; ++t) o[t] = 0.f;
    for (int j = 0; j < 10; ++j) {
        float kv[DHD];
        load64f(V + ((size_t)(row0 + j) * DIM + h * DHD), kv);
        float w = s[j] * inv;
        #pragma unroll
        for (int t = 0; t < DHD; ++t) o[t] += w * kv[t];
    }
    _Float16* op = O + ((size_t)(row0 + i) * DIM + h * DHD);
    #pragma unroll
    for (int qi = 0; qi < 8; ++qi) {
        union { uint4 u; _Float16 h8[8]; } t;
        #pragma unroll
        for (int j = 0; j < 8; ++j) t.h8[j] = (_Float16)o[qi * 8 + j];
        ((uint4*)op)[qi] = t.u;
    }
}

// ================= mean pool over 10 tokens =================
__global__ __launch_bounds__(256) void pool_k(const _Float16* __restrict__ Z,
                                              _Float16* __restrict__ ZP) {
    int idx = blockIdx.x * 256 + threadIdx.x;       // b*512 + d
    if (idx >= BB * DIM) return;
    int b = idx / DIM, d = idx % DIM;
    float s = 0.f;
    #pragma unroll
    for (int t = 0; t < 10; ++t) s += (float)Z[(size_t)(b * 10 + t) * DIM + d];
    ZP[idx] = (_Float16)(s * 0.1f);
}

// ================= final head (N=5, fp32 out) =================
__global__ __launch_bounds__(256) void head2_k(const _Float16* __restrict__ H1,
                                               const float* __restrict__ W2,
                                               const float* __restrict__ b2,
                                               float* __restrict__ out) {
    int idx = blockIdx.x * 256 + threadIdx.x;       // b*5 + nc
    if (idx >= BB * NCL) return;
    int b = idx / NCL, nc = idx % NCL;
    const _Float16* h = H1 + (size_t)b * DIM;
    float s = b2[nc];
    for (int k = 0; k < DIM; ++k) s += (float)h[k] * W2[k * NCL + nc];
    out[idx] = s;
}

// ================= host driver =================
static void launch_gemm(hipStream_t st, const _Float16* A, int lda,
                        const _Float16* W, int ldw, const float* bias,
                        const _Float16* resid, _Float16* C, int ldc,
                        int M, int N, int K, int relu) {
    dim3 g(N / BN, M / BM), b(256);
    gemm_f16_k<false><<<g, b, 0, st>>>(A, lda, W, ldw, bias, resid, C, ldc, K, relu,
                                       nullptr, nullptr, nullptr);
}

extern "C" void kernel_launch(void* const* d_in, const int* in_sizes, int n_in,
                              void* d_out, int out_size, void* d_ws, size_t ws_size,
                              hipStream_t stream) {
    if (ws_size < WS_NEED) return;

    const int*   off_ids  = (const int*)d_in[0];
    const int*   def_ids  = (const int*)d_in[1];
    const int*   off_pos  = (const int*)d_in[2];
    const int*   def_pos  = (const int*)d_in[3];
    const float* playerE  = (const float*)d_in[4];
    const float* roleE    = (const float*)d_in[5];
    const float* proj_W1  = (const float*)d_in[6];
    const float* proj_b1  = (const float*)d_in[7];
    const float* proj_W2  = (const float*)d_in[8];
    const float* proj_b2  = (const float*)d_in[9];
    const float* ln1_g    = (const float*)d_in[10];
    const float* ln1_b    = (const float*)d_in[11];
    const float* Wq       = (const float*)d_in[12];
    const float* Wk       = (const float*)d_in[13];
    const float* Wv       = (const float*)d_in[14];
    const float* Wo       = (const float*)d_in[15];
    const float* bias_B   = (const float*)d_in[16];
    const float* ln2_g    = (const float*)d_in[17];
    const float* ln2_b    = (const float*)d_in[18];
    const float* ffn_W1   = (const float*)d_in[19];
    const float* ffn_b1   = (const float*)d_in[20];
    const float* ffn_W2   = (const float*)d_in[21];
    const float* ffn_b2   = (const float*)d_in[22];
    const float* head_W1  = (const float*)d_in[23];
    const float* head_b1  = (const float*)d_in[24];
    const float* head_W2  = (const float*)d_in[25];
    const float* head_b2  = (const float*)d_in[26];

    _Float16* hw   = (_Float16*)d_ws;
    int*      ip   = (int*)((char*)d_ws + HALF_TOTAL * 2);
    int*      pids = ip;
    int*      cids = ip + NTOK;

    auto cvt = [&](const float* src, size_t off, size_t n) {
        cvt_k<<<dim3((unsigned)((n + 255) / 256)), dim3(256), 0, stream>>>(src, hw + off, (int)n);
    };

    // --- one-time weight conversion to f16 ---
    cvt(playerE, O_PLAYER, (size_t)NPL * DIM);
    cvt(roleE,   O_ROLE,   (size_t)2 * DIM);
    cvt(proj_W1, O_PW1,    (size_t)1024 * 1024);
    cvt(proj_W2, O_PW2,    (size_t)1024 * 512);
    cvt(Wq,      O_WQ,     (size_t)NLAY * DIM * DIM);
    cvt(Wk,      O_WK,     (size_t)NLAY * DIM * DIM);
    cvt(Wv,      O_WV,     (size_t)NLAY * DIM * DIM);
    cvt(Wo,      O_WO,     (size_t)NLAY * DIM * DIM);
    cvt(ffn_W1,  O_F1,     (size_t)NLAY * DIM * FFD);
    cvt(ffn_W2,  O_F2,     (size_t)NLAY * FFD * DIM);
    cvt(head_W1, O_HW1,    (size_t)DIM * DIM);

    prep_k<<<dim3(NTOK / 256), dim3(256), 0, stream>>>(off_ids, def_ids, off_pos,
                                                       def_pos, pids, cids);

    _Float16* Z   = hw + O_Z;
    _Float16* Zn  = hw + O_ZN;
    _Float16* Qb  = hw + O_Q;
    _Float16* Kb  = hw + O_KB;
    _Float16* Vb  = hw + O_VB;

    // --- input projection: P = relu(gather(X) @ W1 + b1); Z = P @ W2 + b2 ---
    // P (81920 x 1024 f16) lives in Q+K scratch.
    {
        dim3 g(1024 / BN, NTOK / BM), b(256);
        gemm_f16_k<true><<<g, b, 0, stream>>>(nullptr, 0, hw + O_PW1, 1024, proj_b1,
                                              nullptr, Qb, 1024, 1024, 1,
                                              pids, hw + O_PLAYER, hw + O_ROLE);
    }
    launch_gemm(stream, Qb, 1024, hw + O_PW2, 512, proj_b2, nullptr, Z, 512,
                NTOK, 512, 1024, 0);

    // --- transformer layers ---
    for (int l = 0; l < NLAY; ++l) {
        const _Float16* wq = hw + O_WQ + (size_t)l * DIM * DIM;
        const _Float16* wk = hw + O_WK + (size_t)l * DIM * DIM;
        const _Float16* wv = hw + O_WV + (size_t)l * DIM * DIM;
        const _Float16* wo = hw + O_WO + (size_t)l * DIM * DIM;
        const _Float16* f1 = hw + O_F1 + (size_t)l * DIM * FFD;
        const _Float16* f2 = hw + O_F2 + (size_t)l * FFD * DIM;

        ln_k<<<dim3(NTOK / 8), dim3(256), 0, stream>>>(Z, ln1_g + l * DIM,
                                                       ln1_b + l * DIM, Zn);
        launch_gemm(stream, Zn, DIM, wq, DIM, nullptr, nullptr, Qb, DIM, NTOK, DIM, DIM, 0);
        launch_gemm(stream, Zn, DIM, wk, DIM, nullptr, nullptr, Kb, DIM, NTOK, DIM, DIM, 0);
        launch_gemm(stream, Zn, DIM, wv, DIM, nullptr, nullptr, Vb, DIM, NTOK, DIM, DIM, 0);

        attn_k<<<dim3((BB * NHD * 10) / 256), dim3(256), 0, stream>>>(
            Qb, Kb, Vb, bias_B + (size_t)l * NHD * 100, cids, Zn);

        // Z = Z + O @ Wo
        launch_gemm(stream, Zn, DIM, wo, DIM, nullptr, Z, Z, DIM, NTOK, DIM, DIM, 0);

        ln_k<<<dim3(NTOK / 8), dim3(256), 0, stream>>>(Z, ln2_g + l * DIM,
                                                       ln2_b + l * DIM, Zn);
        // FFN in 4 M-chunks; hidden (chunk x 2048) reuses Q scratch
        const int MC = NTOK / 4;   // 20480 rows
        for (int c = 0; c < 4; ++c) {
            const _Float16* zn_c = Zn + (size_t)c * MC * DIM;
            _Float16*       z_c  = Z  + (size_t)c * MC * DIM;
            launch_gemm(stream, zn_c, DIM, f1, FFD, ffn_b1 + l * FFD, nullptr,
                        Qb, FFD, MC, FFD, DIM, 1);
            launch_gemm(stream, Qb, FFD, f2, DIM, ffn_b2 + l * DIM, z_c,
                        z_c, DIM, MC, DIM, FFD, 0);
        }
    }

    // --- pooled head ---
    pool_k<<<dim3((BB * DIM) / 256), dim3(256), 0, stream>>>(Z, hw + O_ZP);
    launch_gemm(stream, hw + O_ZP, DIM, hw + O_HW1, DIM, head_b1, nullptr,
                hw + O_H1, DIM, BB, DIM, DIM, 1);
    head2_k<<<dim3((BB * NCL) / 256), dim3(256), 0, stream>>>(hw + O_H1, head_W2,
                                                              head_b2, (float*)d_out);
}
